// FieldAwareFactorizationMachineModel_48945447305506
// MI455X (gfx1250) — matrix operand order repaired and checked by reading, not compiled, not run on hardware
//
#include <hip/hip_runtime.h>
#include <hip/hip_bf16.h>
#include <math.h>

typedef __attribute__((ext_vector_type(16))) _Float16 v16h;
typedef __attribute__((ext_vector_type(8)))  float    v8f;

#define FFM_B 16384
#define FFM_D 1024
#define FFM_F 64
#define FFM_K 64

#define KSPLIT 4               // waves per block, each owns D/KSPLIT columns
#define ZPITCH (FFM_K + 4)     // padded LDS row

// ---------------------------------------------------------------------------
// Phase 1: gather W[i] = v[fi[i], i, :]  (D x K), emit:
//   WhT  : f16, [K][D]  (transposed, so B-fragments are contiguous per lane)
//   swl  : f32, [D][2]  interleaved { s[i] = sum_k W[i,k]^2 , w_lin[i] }
// ---------------------------------------------------------------------------
__global__ __launch_bounds__(256) void ffm_prep(
    const int* __restrict__ fi, const float* __restrict__ v,
    const float* __restrict__ w_lin, _Float16* __restrict__ WhT,
    float* __restrict__ swl) {
  int i = blockIdx.x * blockDim.x + threadIdx.x;
  if (i >= FFM_D) return;
  const float* row = v + ((size_t)fi[i] * FFM_D + (size_t)i) * FFM_K;
  const float4* r4 = (const float4*)row;
  float s = 0.f;
#pragma unroll
  for (int n4 = 0; n4 < FFM_K / 4; ++n4) {
    float4 w = r4[n4];
    s = fmaf(w.x, w.x, s); s = fmaf(w.y, w.y, s);
    s = fmaf(w.z, w.z, s); s = fmaf(w.w, w.w, s);
    int n = n4 * 4;
    WhT[(size_t)(n + 0) * FFM_D + i] = (_Float16)w.x;
    WhT[(size_t)(n + 1) * FFM_D + i] = (_Float16)w.y;
    WhT[(size_t)(n + 2) * FFM_D + i] = (_Float16)w.z;
    WhT[(size_t)(n + 3) * FFM_D + i] = (_Float16)w.w;
  }
  swl[2 * i + 0] = s;
  swl[2 * i + 1] = w_lin[i];
}

// ---------------------------------------------------------------------------
// Phase 2: one block (4 waves) per 16-row tile of x; wave w reduces K-columns
// [256w, 256w+256). Partial z tiles / diag / lin are combined in LDS.
//   z[16,64] = x_tile @ W  via v_wmma_f32_16x16x32_f16 (4 N-tiles per chunk)
//   fused diag = (x*x)@s, lin = x@w_lin on the same x loads
//   out = sigmoid(lin + b + 0.5*(||z||^2 - diag))
// ---------------------------------------------------------------------------
__global__ __launch_bounds__(32 * KSPLIT) void ffm_main(
    const float* __restrict__ x, const _Float16* __restrict__ WhT,
    const float* __restrict__ swl, const float* __restrict__ b_lin,
    float* __restrict__ out) {
  __shared__ float zlds[KSPLIT][16][ZPITCH];  // per-wave partial z tiles
  __shared__ float dlds[KSPLIT][2][16];       // per-wave partial {diag, lin}

  const int tid  = threadIdx.x;
  const int lane = tid & 31;
  const int wv   = tid >> 5;      // wave id within block: 0..KSPLIT-1
  const int c    = lane & 15;     // column id / row-in-tile id
  const int hi   = lane >> 4;     // half-wave select
  const int row0 = blockIdx.x * 16;

  const float* xrow = x + (size_t)(row0 + c) * FFM_D;
  const int kbeg = wv * (FFM_D / KSPLIT);
  const int kend = kbeg + (FFM_D / KSPLIT);

  v8f acc0 = {}, acc1 = {}, acc2 = {}, acc3 = {};
  float diag = 0.f, lin = 0.f;

  for (int kc = kbeg; kc < kend; kc += 32) {
    const int k0 = kc + 16 * hi;       // A-frag halves h=0..7  -> K = k0 + h
    const int k1 = kc + 16 * hi + 8;   // A-frag halves h=8..15 -> K = k1 + (h-8)

    // x tile: each (m,k) element of the 16x32 chunk loaded exactly once.
    float4 xa = *(const float4*)(xrow + k0);
    float4 xb = *(const float4*)(xrow + k0 + 4);
    float4 xc = *(const float4*)(xrow + k1);
    float4 xd = *(const float4*)(xrow + k1 + 4);

    // interleaved {s, w_lin} pairs (broadcast across 16 lanes, L2/L0-hot)
    float4 p0 = *(const float4*)(swl + 2 * k0);
    float4 p1 = *(const float4*)(swl + 2 * k0 + 4);
    float4 p2 = *(const float4*)(swl + 2 * k0 + 8);
    float4 p3 = *(const float4*)(swl + 2 * k0 + 12);
    float4 q0 = *(const float4*)(swl + 2 * k1);
    float4 q1 = *(const float4*)(swl + 2 * k1 + 4);
    float4 q2 = *(const float4*)(swl + 2 * k1 + 8);
    float4 q3 = *(const float4*)(swl + 2 * k1 + 12);

    // fused diag/linear accumulation on the f32 values (co-executes with WMMA)
    lin  = fmaf(xa.x, p0.y, lin); diag = fmaf(xa.x * xa.x, p0.x, diag);
    lin  = fmaf(xa.y, p0.w, lin); diag = fmaf(xa.y * xa.y, p0.z, diag);
    lin  = fmaf(xa.z, p1.y, lin); diag = fmaf(xa.z * xa.z, p1.x, diag);
    lin  = fmaf(xa.w, p1.w, lin); diag = fmaf(xa.w * xa.w, p1.z, diag);
    lin  = fmaf(xb.x, p2.y, lin); diag = fmaf(xb.x * xb.x, p2.x, diag);
    lin  = fmaf(xb.y, p2.w, lin); diag = fmaf(xb.y * xb.y, p2.z, diag);
    lin  = fmaf(xb.z, p3.y, lin); diag = fmaf(xb.z * xb.z, p3.x, diag);
    lin  = fmaf(xb.w, p3.w, lin); diag = fmaf(xb.w * xb.w, p3.z, diag);
    lin  = fmaf(xc.x, q0.y, lin); diag = fmaf(xc.x * xc.x, q0.x, diag);
    lin  = fmaf(xc.y, q0.w, lin); diag = fmaf(xc.y * xc.y, q0.z, diag);
    lin  = fmaf(xc.z, q1.y, lin); diag = fmaf(xc.z * xc.z, q1.x, diag);
    lin  = fmaf(xc.w, q1.w, lin); diag = fmaf(xc.w * xc.w, q1.z, diag);
    lin  = fmaf(xd.x, q2.y, lin); diag = fmaf(xd.x * xd.x, q2.x, diag);
    lin  = fmaf(xd.y, q2.w, lin); diag = fmaf(xd.y * xd.y, q2.z, diag);
    lin  = fmaf(xd.z, q3.y, lin); diag = fmaf(xd.z * xd.z, q3.x, diag);
    lin  = fmaf(xd.w, q3.w, lin); diag = fmaf(xd.w * xd.w, q3.z, diag);

    // Build A fragment (lane=M, halves map to K above, same k order as B)
    v16h a;
    a[0]  = (_Float16)xa.x; a[1]  = (_Float16)xa.y;
    a[2]  = (_Float16)xa.z; a[3]  = (_Float16)xa.w;
    a[4]  = (_Float16)xb.x; a[5]  = (_Float16)xb.y;
    a[6]  = (_Float16)xb.z; a[7]  = (_Float16)xb.w;
    a[8]  = (_Float16)xc.x; a[9]  = (_Float16)xc.y;
    a[10] = (_Float16)xc.z; a[11] = (_Float16)xc.w;
    a[12] = (_Float16)xd.x; a[13] = (_Float16)xd.y;
    a[14] = (_Float16)xd.z; a[15] = (_Float16)xd.w;

    // B fragments from transposed W (lane=N, half h -> K=16*hi+h, same k order as A)
    const _Float16* bp = WhT + kc + 16 * hi;
    v16h b0 = *(const v16h*)(bp + (size_t)(c +  0) * FFM_D);
    v16h b1 = *(const v16h*)(bp + (size_t)(c + 16) * FFM_D);
    v16h b2 = *(const v16h*)(bp + (size_t)(c + 32) * FFM_D);
    v16h b3 = *(const v16h*)(bp + (size_t)(c + 48) * FFM_D);

    acc0 = __builtin_amdgcn_wmma_f32_16x16x32_f16(false, a, false, b0, (short)0, acc0, false, false);
    acc1 = __builtin_amdgcn_wmma_f32_16x16x32_f16(false, a, false, b1, (short)0, acc1, false, false);
    acc2 = __builtin_amdgcn_wmma_f32_16x16x32_f16(false, a, false, b2, (short)0, acc2, false, false);
    acc3 = __builtin_amdgcn_wmma_f32_16x16x32_f16(false, a, false, b3, (short)0, acc3, false, false);
  }

  // Scatter partial z to this wave's LDS copy.
  // C/D 16x16 f32 layout: VGPR r holds M = r + 8*hi, lane holds N = c.
#pragma unroll
  for (int r = 0; r < 8; ++r) {
    int m = r + 8 * hi;
    zlds[wv][m][ 0 + c] = acc0[r];
    zlds[wv][m][16 + c] = acc1[r];
    zlds[wv][m][32 + c] = acc2[r];
    zlds[wv][m][48 + c] = acc3[r];
  }

  // Combine the two hi-halves of this wave's diag/lin (disjoint k coverage)
  diag += __shfl_xor(diag, 16, 32);
  lin  += __shfl_xor(lin, 16, 32);
  if (lane < 16) {
    dlds[wv][0][c] = diag;
    dlds[wv][1][c] = lin;
  }
  __syncthreads();

  // Final reduction: 16 threads, one per output row.
  if (tid < 16) {
    const float4* z0 = (const float4*)&zlds[0][tid][0];
    const float4* z1 = (const float4*)&zlds[1][tid][0];
    const float4* z2 = (const float4*)&zlds[2][tid][0];
    const float4* z3 = (const float4*)&zlds[3][tid][0];
    float full = 0.f;
#pragma unroll
    for (int j = 0; j < FFM_K / 4; ++j) {
      float4 a0 = z0[j], a1 = z1[j], a2 = z2[j], a3 = z3[j];
      float zx = (a0.x + a1.x) + (a2.x + a3.x);
      float zy = (a0.y + a1.y) + (a2.y + a3.y);
      float zz = (a0.z + a1.z) + (a2.z + a3.z);
      float zw = (a0.w + a1.w) + (a2.w + a3.w);
      full = fmaf(zx, zx, full); full = fmaf(zy, zy, full);
      full = fmaf(zz, zz, full); full = fmaf(zw, zw, full);
    }
    float dsum = (dlds[0][0][tid] + dlds[1][0][tid]) +
                 (dlds[2][0][tid] + dlds[3][0][tid]);
    float lsum = (dlds[0][1][tid] + dlds[1][1][tid]) +
                 (dlds[2][1][tid] + dlds[3][1][tid]);
    float t = lsum + b_lin[0] + 0.5f * (full - dsum);
    out[row0 + tid] = 1.0f / (1.0f + __expf(-t));
  }
}

// ---------------------------------------------------------------------------
extern "C" void kernel_launch(void* const* d_in, const int* in_sizes, int n_in,
                              void* d_out, int out_size, void* d_ws, size_t ws_size,
                              hipStream_t stream) {
  const float* x     = (const float*)d_in[0];   // [B, D]
  const int*   fi    = (const int*)  d_in[1];   // [D]
  const float* v     = (const float*)d_in[2];   // [F, D, K]
  const float* w_lin = (const float*)d_in[3];   // [D]
  const float* b_lin = (const float*)d_in[4];   // [1]
  float* out = (float*)d_out;                   // [B]

  _Float16* WhT = (_Float16*)d_ws;                               // 128 KB
  float*    swl = (float*)((char*)d_ws + (size_t)FFM_K * FFM_D * sizeof(_Float16)); // 8 KB

  ffm_prep<<<FFM_D / 256, 256, 0, stream>>>(fi, v, w_lin, WhT, swl);
  ffm_main<<<FFM_B / 16, 32 * KSPLIT, 0, stream>>>(x, WhT, swl, b_lin, out);
}
